// GeoVisFusion_8211977470292
// MI455X (gfx1250) — matrix operand order repaired and checked by reading, not compiled, not run on hardware
//
#include <hip/hip_runtime.h>
#include <hip/hip_bf16.h>

typedef __attribute__((ext_vector_type(16))) _Float16 v16h;
typedef __attribute__((ext_vector_type(8)))  _Float16 v8h;
typedef __attribute__((ext_vector_type(8)))  float    v8f;

#define NQ      65536
#define MV      1558
#define NUMV    779
#define WAVES   4
#define FSTR    232   // fused staging stride (halves): 464B/row, 16B aligned, spreads banks
#define H1STR   72    // hidden-64 stride
#define HIDSTR  24    // hidden-16 stride

// packed-weight fragment offsets (in halves); each frag = 512 halves = 1KB
#define OFF_A1   0      // 224x16  (7 frags)
#define OFF_A2   3584   // 32x16   (1)
#define OFF_B1   4096   // 224x64  (28)
#define OFF_B2   18432  // 64x64   (8)
#define OFF_A11  22528  // 32x16
#define OFF_A21  23040  // 32x16
#define OFF_B11  23552  // 32x16
#define OFF_B21  24064  // 32x16
#define WP_TOTAL 24576

// ---------------- KNN: brute force, vertices cached in LDS ----------------
__global__ __launch_bounds__(256) void knn_kernel(const float* __restrict__ v,
                                                  const float* __restrict__ vert,
                                                  int* __restrict__ idx) {
    __shared__ float sv[MV * 3];
    for (int i = threadIdx.x; i < MV * 3; i += 256) sv[i] = vert[i];
    __syncthreads();
    int n = blockIdx.x * 256 + threadIdx.x;
    float qx = v[n * 3 + 0], qy = v[n * 3 + 1], qz = v[n * 3 + 2];
    float best = 3.4e38f;
    int bi = 0;
    for (int m = 0; m < MV; ++m) {
        float dx = qx - sv[m * 3 + 0];
        float dy = qy - sv[m * 3 + 1];
        float dz = qz - sv[m * 3 + 2];
        float d = dx * dx + dy * dy + dz * dz;
        if (d < best) { best = d; bi = m; }
    }
    idx[n] = bi;
}

// ------- bilinear sample of [C,256,256] grid at vertex uv, times vis -------
__global__ __launch_bounds__(256) void sample_kernel(const float* __restrict__ feat, int C,
                                                     const float* __restrict__ vert_xy,
                                                     const float* __restrict__ vis,
                                                     float* __restrict__ out, int total) {
    int t = blockIdx.x * 256 + threadIdx.x;
    if (t >= total) return;
    int m = t / C, c = t % C;
    float u = vert_xy[m * 2 + 0], w = vert_xy[m * 2 + 1];
    float x = (u + 1.f) * 0.5f * 255.f;
    float y = (w + 1.f) * 0.5f * 255.f;
    float x0f = floorf(x), y0f = floorf(y);
    float wx = x - x0f, wy = y - y0f;
    int xi0 = min(max((int)x0f, 0), 255);
    int xi1 = min(max((int)x0f + 1, 0), 255);
    int yi0 = min(max((int)y0f, 0), 255);
    int yi1 = min(max((int)y0f + 1, 0), 255);
    const float* fc = feat + c * 65536;
    float f00 = fc[yi0 * 256 + xi0], f01 = fc[yi0 * 256 + xi1];
    float f10 = fc[yi1 * 256 + xi0], f11 = fc[yi1 * 256 + xi1];
    float val = f00 * (1.f - wx) * (1.f - wy) + f01 * wx * (1.f - wy)
              + f10 * (1.f - wx) * wy + f11 * wx * wy;
    out[t] = val * vis[m];
}

// ---- pack W[O][I] (row-major) into f16 B-fragments, fragment-major order ----
// fragment f=(kt*(Npad/16)+nt): element (lane,i) = W[nt*16+(lane&15)][kt*32+((i>>3)<<4)+((lane>>4)<<3)+(i&7)]
__device__ inline void packW(const float* __restrict__ W, int O, int I,
                             _Float16* __restrict__ dst, int Kpad, int Npad) {
    int total = Kpad * Npad;
    int nf = Npad >> 4;
    for (int e = threadIdx.x; e < total; e += 256) {
        int frag = e >> 9, w = e & 511, lane = w >> 4, i = w & 15;
        int kt = frag / nf, nt = frag % nf;
        int k = kt * 32 + ((i >> 3) << 4) + ((lane >> 4) << 3) + (i & 7);
        int n = nt * 16 + (lane & 15);
        float val = (k < I && n < O) ? W[n * I + k] : 0.f;
        dst[e] = (_Float16)val;
    }
}

__global__ __launch_bounds__(256) void pack_kernel(
    const float* Wa1, const float* Wa2, const float* Wb1, const float* Wb2,
    const float* Wa11, const float* Wa21, const float* Wb11, const float* Wb21,
    _Float16* __restrict__ dst) {
    packW(Wa1,  10, 196, dst + OFF_A1,  224, 16);
    packW(Wa2,   3,  10, dst + OFF_A2,   32, 16);
    packW(Wb1,  64, 196, dst + OFF_B1,  224, 64);
    packW(Wb2,  64,  64, dst + OFF_B2,   64, 64);
    packW(Wa11, 10,  28, dst + OFF_A11,  32, 16);
    packW(Wa21,  3,  10, dst + OFF_A21,  32, 16);
    packW(Wb11,  8,  28, dst + OFF_B11,  32, 16);
    packW(Wb21,  8,   8, dst + OFF_B21,  32, 16);
}

// ---------------- WMMA fragment helpers (wave32 layouts per ISA 7.12.2) ----------------
__device__ inline v8f wmma_f16(v16h a, v16h b, v8f c) {
    return __builtin_amdgcn_wmma_f32_16x16x32_f16(false, a, false, b, (short)0, c, false, false);
}

// A 16x32 f16: lane L -> row M=L&15; halves i<8: K=kbase+8*(L>>4)+i ; i>=8: +16.
// Both 8-half runs are contiguous -> two ds_load_b128.
__device__ inline v16h load_A(const _Float16* __restrict__ lds, int stride, int kbase) {
    int lane = threadIdx.x & 31;
    const _Float16* p = lds + (lane & 15) * stride + kbase + ((lane >> 4) << 3);
    v8h lo = *(const v8h*)p;
    v8h hi = *(const v8h*)(p + 16);
    return __builtin_shufflevector(lo, hi, 0,1,2,3,4,5,6,7,8,9,10,11,12,13,14,15);
}

// A fragment where only K=0..15 are valid (hidden-16 layers); K>=16 forced to 0.
__device__ inline v16h load_A_k16(const _Float16* __restrict__ lds, int stride) {
    int lane = threadIdx.x & 31;
    const _Float16* p = lds + (lane & 15) * stride + ((lane >> 4) << 3);
    v8h lo = *(const v8h*)p;
    v8h hi = (v8h)(_Float16)0.0f;
    return __builtin_shufflevector(lo, hi, 0,1,2,3,4,5,6,7,8,9,10,11,12,13,14,15);
}

// B 32x16 f16 fragment, fragment-major in global: lane reads 16 contiguous halves (32B).
__device__ inline v16h load_Bfrag(const _Float16* __restrict__ g) {
    int lane = threadIdx.x & 31;
    return *(const v16h*)(g + lane * 16);
}

// C/D 16x16 f32: lane L -> col N=L&15, rows M=(L>>4)*8 + vr.
__device__ inline void store_C_relu(v8f acc, _Float16* __restrict__ lds, int stride, int nbase) {
    int n = threadIdx.x & 15, m0 = ((threadIdx.x >> 4) & 1) * 8;
    #pragma unroll
    for (int vr = 0; vr < 8; ++vr)
        lds[(m0 + vr) * stride + nbase + n] = (_Float16)fmaxf(acc[vr], 0.f);
}

// ---------------- Fused dual-branch gated MLP ----------------
__global__ __launch_bounds__(128) void fusion_kernel(
    const float* __restrict__ fs0, const float* __restrict__ fs1,
    const float* __restrict__ sdf, const float* __restrict__ qvis,
    const float* __restrict__ vvis, const int* __restrict__ knn,
    const float* __restrict__ vfv0, const float* __restrict__ vfv1,
    const _Float16* __restrict__ wp,
    float* __restrict__ out0, float* __restrict__ out1) {
    __shared__ _Float16 fusedS[WAVES][16 * FSTR];
    __shared__ _Float16 h1S[WAVES][16 * H1STR];
    __shared__ _Float16 hidS[WAVES][16 * HIDSTR];
    __shared__ float    atS[WAVES][64];

    const int lane = threadIdx.x & 31;
    const int wave = threadIdx.x >> 5;
    const int rowBase = (blockIdx.x * WAVES + wave) * 16;

    _Float16* fused = fusedS[wave];
    _Float16* h1    = h1S[wave];
    _Float16* hid   = hidS[wave];
    float*    atb   = atS[wave];

    const int r   = lane & 15;   // staging row served by this lane
    const int par = lane >> 4;   // k parity
    const int row = rowBase + r;
    const int ki  = knn[row];
    const int kto = (ki + NUMV) % (2 * NUMV);
    const float s_sdf = sdf[row];
    const float s_qv  = qvis[row];
    const float s_vth = vvis[ki];
    const float s_vto = vvis[kto];
    const int nn = lane & 15, m0 = (lane >> 4) * 8;

    // ---- build fused [16 x 224] (196 real + zero pad) ----
    for (int k = par; k < 224; k += 2) {
        float val;
        if (k < 64)        val = fs0[row * 64 + k];
        else if (k < 128)  val = vfv0[ki  * 64 + (k - 64)];
        else if (k < 192)  val = vfv0[kto * 64 + (k - 128)];
        else if (k == 192) val = s_sdf;
        else if (k == 193) val = s_qv;
        else if (k == 194) val = s_vth;
        else if (k == 195) val = s_vto;
        else               val = 0.f;
        fused[r * FSTR + k] = (_Float16)val;
    }
    __syncthreads();

    // ---- gate layer A1: [16x224]@[224x16], relu ----
    v8f acc = {};
    #pragma unroll
    for (int kt = 0; kt < 7; ++kt)
        acc = wmma_f16(load_A(fused, FSTR, kt * 32), load_Bfrag(wp + OFF_A1 + kt * 512), acc);
    store_C_relu(acc, hid, HIDSTR, 0);
    __syncthreads();

    // ---- gate layer A2: [16x16pad32]@[32x16], sigmoid, keep cols 0..2 ----
    {
        v8f z = {};
        acc = wmma_f16(load_A_k16(hid, HIDSTR), load_Bfrag(wp + OFF_A2), z);
    }
    if (nn < 3) {
        #pragma unroll
        for (int vr = 0; vr < 8; ++vr)
            atb[(m0 + vr) * 4 + nn] = 1.f / (1.f + __expf(-acc[vr]));
    }
    __syncthreads();

    // ---- scale fused segments in place by gate ----
    {
        float a0 = atb[r * 4 + 0], a1 = atb[r * 4 + 1], a2 = atb[r * 4 + 2];
        for (int k = par; k < 192; k += 2) {
            float s = (k < 64) ? a0 : (k < 128) ? a1 : a2;
            fused[r * FSTR + k] = (_Float16)((float)fused[r * FSTR + k] * s);
        }
    }
    __syncthreads();

    // ---- B1: [16x224]@[224x64], relu ----
    #pragma unroll
    for (int nt = 0; nt < 4; ++nt) {
        v8f a1c = {};
        #pragma unroll
        for (int kt = 0; kt < 7; ++kt)
            a1c = wmma_f16(load_A(fused, FSTR, kt * 32),
                           load_Bfrag(wp + OFF_B1 + (kt * 4 + nt) * 512), a1c);
        store_C_relu(a1c, h1, H1STR, nt * 16);
    }
    __syncthreads();

    // ---- B2: [16x64]@[64x64] -> out0 ----
    #pragma unroll
    for (int nt = 0; nt < 4; ++nt) {
        v8f a2c = {};
        #pragma unroll
        for (int kt = 0; kt < 2; ++kt)
            a2c = wmma_f16(load_A(h1, H1STR, kt * 32),
                           load_Bfrag(wp + OFF_B2 + (kt * 4 + nt) * 512), a2c);
        #pragma unroll
        for (int vr = 0; vr < 8; ++vr)
            out0[(rowBase + m0 + vr) * 64 + nt * 16 + nn] = a2c[vr];
    }
    __syncthreads();

    // ================= branch 1 (8-dim features, K pad 32) =================
    for (int k = par; k < 32; k += 2) {
        float val;
        if (k < 8)        val = fs1[row * 8 + k];
        else if (k < 16)  val = vfv1[ki  * 8 + (k - 8)];
        else if (k < 24)  val = vfv1[kto * 8 + (k - 16)];
        else if (k == 24) val = s_sdf;
        else if (k == 25) val = s_qv;
        else if (k == 26) val = s_vth;
        else if (k == 27) val = s_vto;
        else              val = 0.f;
        fused[r * FSTR + k] = (_Float16)val;
    }
    __syncthreads();

    { v8f z = {}; acc = wmma_f16(load_A(fused, FSTR, 0), load_Bfrag(wp + OFF_A11), z); }
    store_C_relu(acc, hid, HIDSTR, 0);
    __syncthreads();

    { v8f z = {}; acc = wmma_f16(load_A_k16(hid, HIDSTR), load_Bfrag(wp + OFF_A21), z); }
    if (nn < 3) {
        #pragma unroll
        for (int vr = 0; vr < 8; ++vr)
            atb[(m0 + vr) * 4 + nn] = 1.f / (1.f + __expf(-acc[vr]));
    }
    __syncthreads();

    {
        float a0 = atb[r * 4 + 0], a1 = atb[r * 4 + 1], a2 = atb[r * 4 + 2];
        for (int k = par; k < 24; k += 2) {
            float s = (k < 8) ? a0 : (k < 16) ? a1 : a2;
            fused[r * FSTR + k] = (_Float16)((float)fused[r * FSTR + k] * s);
        }
    }
    __syncthreads();

    { v8f z = {}; acc = wmma_f16(load_A(fused, FSTR, 0), load_Bfrag(wp + OFF_B11), z); }
    store_C_relu(acc, hid, HIDSTR, 0);
    __syncthreads();

    { v8f z = {}; acc = wmma_f16(load_A_k16(hid, HIDSTR), load_Bfrag(wp + OFF_B21), z); }
    if (nn < 8) {
        #pragma unroll
        for (int vr = 0; vr < 8; ++vr)
            out1[(rowBase + m0 + vr) * 8 + nn] = acc[vr];
    }
}

extern "C" void kernel_launch(void* const* d_in, const int* in_sizes, int n_in,
                              void* d_out, int out_size, void* d_ws, size_t ws_size,
                              hipStream_t stream) {
    const float* vert_xy = (const float*)d_in[0];
    const float* fg0     = (const float*)d_in[1];
    const float* fg1     = (const float*)d_in[2];
    const float* fs0     = (const float*)d_in[3];
    const float* fs1     = (const float*)d_in[4];
    const float* vert    = (const float*)d_in[5];
    const float* v       = (const float*)d_in[6];
    const float* vvis    = (const float*)d_in[7];
    const float* qvis    = (const float*)d_in[8];
    // d_in[9] = closest_face (unused by reference outputs)
    const float* qsdf    = (const float*)d_in[10];
    const float* Wa1  = (const float*)d_in[11];
    const float* Wa2  = (const float*)d_in[12];
    const float* Wb1  = (const float*)d_in[13];
    const float* Wb2  = (const float*)d_in[14];
    const float* Wa11 = (const float*)d_in[15];
    const float* Wa21 = (const float*)d_in[16];
    const float* Wb11 = (const float*)d_in[17];
    const float* Wb21 = (const float*)d_in[18];

    char* ws = (char*)d_ws;
    int*       idx  = (int*)ws;                                   // 65536 ints
    float*     vfv0 = (float*)(ws + NQ * sizeof(int));            // 1558*64 f32
    float*     vfv1 = vfv0 + MV * 64;                             // 1558*8  f32
    _Float16*  wp   = (_Float16*)(vfv1 + MV * 8);                 // 24576 halves (32B-aligned)

    knn_kernel<<<NQ / 256, 256, 0, stream>>>(v, vert, idx);
    sample_kernel<<<(MV * 64 + 255) / 256, 256, 0, stream>>>(fg0, 64, vert_xy, vvis, vfv0, MV * 64);
    sample_kernel<<<(MV * 8 + 255) / 256, 256, 0, stream>>>(fg1, 8, vert_xy, vvis, vfv1, MV * 8);
    pack_kernel<<<1, 256, 0, stream>>>(Wa1, Wa2, Wb1, Wb2, Wa11, Wa21, Wb11, Wb21, wp);

    float* out0 = (float*)d_out;
    float* out1 = out0 + (size_t)NQ * 64;
    fusion_kernel<<<(NQ / 16) / WAVES, WAVES * 32, 0, stream>>>(
        fs0, fs1, qsdf, qvis, vvis, idx, vfv0, vfv1, wp, out0, out1);
}
